// AttentionBlock3D_34789235098215
// MI455X (gfx1250) — matrix operand order repaired and checked by reading, not compiled, not run on hardware
//
#include <hip/hip_runtime.h>
#include <hip/hip_bf16.h>

// AttentionBlock3D for MI455X (gfx1250, wave32, WMMA).
// b=2, c=256, n=16^3=4096, heads=8, hdim=64, inner=512.
// Compute-bound (~77 GFLOP vs ~40MB data, all L2-resident) -> all GEMMs via
// v_wmma_f32_16x16x32_f16 (f16 in, f32 accumulate). Attention K/V tiles are
// staged into LDS with double-buffered GLOBAL_LOAD_ASYNC_TO_LDS_B128
// (ASYNCcnt-tracked), overlapping the copy with WMMA + softmax.

#define B_   2
#define C_   256
#define N_   4096
#define HEADS_ 8
#define HDIM_  64
#define INNER_ 512
#define QKV_ROWS_ 1536
#define SCALE_ 0.125f
#define EPS_ 1e-5f

typedef __attribute__((ext_vector_type(16))) _Float16 v16h;
typedef __attribute__((ext_vector_type(8)))  float    v8f;

__device__ __forceinline__ v8f wmma16(v16h a, v16h b, v8f c) {
  return __builtin_amdgcn_wmma_f32_16x16x32_f16(false, a, false, b, (short)0, c, false, false);
}

// Async copy 16 bytes global -> LDS (per active lane). LDS operand is the
// wave-relative LDS byte offset = low 32 bits of the generic pointer
// (shared aperture lives entirely in addr[63:32]).
__device__ __forceinline__ void async_ld_b128(const _Float16* gsrc, _Float16* ldst) {
  asm volatile("global_load_async_to_lds_b128 %0, %1, off"
               :: "v"((unsigned int)(unsigned long long)ldst), "v"(gsrc)
               : "memory");
}

#if __has_builtin(__builtin_amdgcn_s_wait_asynccnt)
#define WAIT_ASYNC(n) __builtin_amdgcn_s_wait_asynccnt(n)
#else
#define WAIT_ASYNC(n) asm volatile("s_wait_asynccnt %0" :: "i"(n) : "memory")
#endif

// A-matrix fragment (16x32 f16), src row-major [M x ld], tile at (row0, k0).
// Lane holds row M = lane%16; VGPR v holds K = 16*hi + 2v, 2v+1 (matches load_b_frag).
__device__ __forceinline__ v16h load_a_frag(const _Float16* src, int ld, int row0, int k0) {
  int lane = threadIdx.x & 31;
  int row  = row0 + (lane & 15);
  int hi   = lane >> 4;
  const _Float16* base = src + (size_t)row * ld + k0;
  v16h a;
#pragma unroll
  for (int v = 0; v < 8; ++v) {
    int k = 2 * v + 16 * hi;
    a[2 * v]     = base[k];
    a[2 * v + 1] = base[k + 1];
  }
  return a;
}

// B-matrix fragment (32x16 f16). Element (k,n) stored at src[(col0+n)*ld + k0 + k]
// (i.e. B is stored N-major with K contiguous). Lane holds col N = lane%16;
// VGPR v holds K = 16*hi + 2v, 2v+1.
__device__ __forceinline__ v16h load_b_frag(const _Float16* src, int ld, int k0, int col0) {
  int lane = threadIdx.x & 31;
  int col  = col0 + (lane & 15);
  int hi   = lane >> 4;
  const _Float16* p = src + (size_t)col * ld + k0 + 16 * hi;
  v16h b;
#pragma unroll
  for (int v = 0; v < 8; ++v) {
    b[2 * v]     = p[2 * v];
    b[2 * v + 1] = p[2 * v + 1];
  }
  return b;
}

// C/D tile (16x16 f32): VGPR r holds row r + 8*hi, col = lane%16. Store as f16.
__device__ __forceinline__ void store_c_f16(_Float16* dst, int ld, const v8f& c) {
  int lane = threadIdx.x & 31;
  int col  = lane & 15;
  int hi   = lane >> 4;
#pragma unroll
  for (int r = 0; r < 8; ++r) dst[(r + 8 * hi) * ld + col] = (_Float16)c[r];
}

// ---------------- Kernel 1: GroupNorm stats ----------------
// grid = 16 blocks, one per (b, group). Each group = 32 channels x 4096 = 131072 floats,
// contiguous in x[b, g*32 .. g*32+31, :].
__global__ void gn_stats_kernel(const float* __restrict__ x, float* __restrict__ stats) {
  __shared__ float sS[256], sQ[256];
  const int bg = blockIdx.x;                 // 0..15
  const size_t base = (size_t)bg * 32 * N_;  // (b*256 + g*32)*4096
  float s = 0.f, q = 0.f;
  for (int i = threadIdx.x; i < 32 * N_; i += 256) {
    float v = x[base + i];
    s += v; q += v * v;
  }
  sS[threadIdx.x] = s; sQ[threadIdx.x] = q;
  __syncthreads();
  for (int off = 128; off > 0; off >>= 1) {
    if (threadIdx.x < off) { sS[threadIdx.x] += sS[threadIdx.x + off]; sQ[threadIdx.x] += sQ[threadIdx.x + off]; }
    __syncthreads();
  }
  if (threadIdx.x == 0) {
    const float inv_n = 1.0f / (32.0f * N_);
    float mu  = sS[0] * inv_n;
    float var = sQ[0] * inv_n - mu * mu;
    stats[bg * 2]     = mu;
    stats[bg * 2 + 1] = rsqrtf(var + EPS_);
  }
}

// ---------------- Kernel 2: GN apply -> xnT [b, n, c] f16 ----------------
__global__ void gn_apply_kernel(const float* __restrict__ x, const float* __restrict__ gamma,
                                const float* __restrict__ beta, const float* __restrict__ stats,
                                _Float16* __restrict__ xnT) {
  const int bn = blockIdx.x;            // 0 .. 2*4096-1
  const int b = bn >> 12, n = bn & 4095;
  const int c = threadIdx.x;            // 0..255
  const int g = c >> 5;
  const float mu   = stats[(b * 8 + g) * 2];
  const float rinv = stats[(b * 8 + g) * 2 + 1];
  const float v = (x[((size_t)(b * C_ + c)) * N_ + n] - mu) * rinv * gamma[c] + beta[c];
  xnT[((size_t)(b * N_ + n)) * C_ + c] = (_Float16)v;
}

// ---------------- Kernel 3: weight conversion ----------------
__global__ void convert_w_kernel(const float* __restrict__ w_qkv, const float* __restrict__ w_out,
                                 _Float16* __restrict__ w_h, _Float16* __restrict__ wout_h) {
  const int i = blockIdx.x * 256 + threadIdx.x;
  const int NQ = QKV_ROWS_ * C_;        // 393216
  const int NO = C_ * INNER_;           // 131072
  if (i < NQ)            w_h[i]      = (_Float16)w_qkv[i];
  else if (i < NQ + NO)  wout_h[i - NQ] = (_Float16)w_out[i - NQ];
}

// ---------------- Kernel 4: QKV GEMM ----------------
// qkv[o, n] = sum_c w[o, c] * xnT[n, c], o in [0,1536).  One 16x16 tile per wave.
// grid = (96 o-tiles, 32 n-chunks of 128, b=2); block = 256 threads = 8 waves.
__global__ void qkv_gemm_kernel(const _Float16* __restrict__ w_h, const _Float16* __restrict__ xnT,
                                const float* __restrict__ b_qkv,
                                _Float16* __restrict__ qbuf, _Float16* __restrict__ kbuf,
                                _Float16* __restrict__ vT) {
  const int wave = threadIdx.x >> 5;
  const int lane = threadIdx.x & 31;
  const int b  = blockIdx.z;
  const int o0 = blockIdx.x * 16;
  const int n0 = blockIdx.y * 128 + wave * 16;
  const _Float16* xb = xnT + (size_t)b * N_ * C_;

  v8f c = {};
#pragma unroll
  for (int kk = 0; kk < 8; ++kk)
    c = wmma16(load_a_frag(w_h, C_, o0, kk * 32), load_b_frag(xb, C_, kk * 32, n0), c);

  const int col = lane & 15, hi = lane >> 4;
#pragma unroll
  for (int r = 0; r < 8; ++r) {
    const int o = o0 + r + 8 * hi;
    const int n = n0 + col;
    const float val = c[r] + b_qkv[o];
    const int which = o >> 9, rem = o & 511, h = rem >> 6, dd = rem & 63;
    const _Float16 hv = (_Float16)val;
    const size_t bh = (size_t)(b * HEADS_ + h);
    if (which == 0)      qbuf[(bh * N_ + n) * HDIM_ + dd] = hv;
    else if (which == 1) kbuf[(bh * N_ + n) * HDIM_ + dd] = hv;
    else                 vT[(bh * HDIM_ + dd) * N_ + n]   = hv;   // transposed for PV B-operand
  }
}

// ---------------- Kernel 5: flash attention ----------------
// grid = (32 n-chunks of 128 q-rows, 16 bh); block = 256 = 8 waves, one 16-row q tile each.
// Key blocks of 32 double-buffered in LDS via async global->LDS copies (ASYNCcnt);
// online softmax in f32; O accumulated in f32.
__global__ void attention_kernel(const _Float16* __restrict__ qbuf, const _Float16* __restrict__ kbuf,
                                 const _Float16* __restrict__ vT, _Float16* __restrict__ obuf) {
  __shared__ __align__(16) _Float16 kS[2][32 * 64];    // [buf][key][d]
  __shared__ __align__(16) _Float16 vS[2][64 * 32];    // [buf][d][key]
  __shared__ __align__(16) _Float16 pS[8 * 16 * 32];   // per-wave P scratch

  const int tid  = threadIdx.x;
  const int wave = tid >> 5;
  const int lane = tid & 31;
  const int bh = blockIdx.y;               // b*8 + h
  const int b = bh >> 3, h = bh & 7;
  const int q0 = blockIdx.x * 128 + wave * 16;
  const int NB = N_ / 32;                  // 128 key blocks

  const _Float16* qb = qbuf + (size_t)bh * N_ * HDIM_;
  const _Float16* kb = kbuf + (size_t)bh * N_ * HDIM_;
  const _Float16* vb = vT   + (size_t)bh * HDIM_ * N_;

  const int dRow = tid >> 2, part = tid & 3;  // V staging: 64 rows x 4 x 16B

  // stage key-block `blk` into LDS buffer `buf`: 2 async b128 per thread
  auto stage = [&](int blk, int buf) {
    const _Float16* ksrc = kb + (size_t)blk * 32 * HDIM_ + tid * 8;
    async_ld_b128(ksrc, &kS[buf][tid * 8]);
    const _Float16* vsrc = vb + (size_t)dRow * N_ + blk * 32 + part * 8;
    async_ld_b128(vsrc, &vS[buf][dRow * 32 + part * 8]);
  };

  const v16h qa0 = load_a_frag(qb, HDIM_, q0, 0);
  const v16h qa1 = load_a_frag(qb, HDIM_, q0, 32);

  v8f o0t = {}, o1t = {}, o2t = {}, o3t = {};
  float m[8], l[8];
#pragma unroll
  for (int r = 0; r < 8; ++r) { m[r] = -3.0e38f; l[r] = 0.f; }

  _Float16* pw = pS + wave * (16 * 32);

  stage(0, 0);
  for (int kbk = 0; kbk < NB; ++kbk) {
    if (kbk + 1 < NB) {
      stage(kbk + 1, (kbk + 1) & 1);       // prefetch next block (async)
      __builtin_prefetch(kb + (size_t)(kbk + 2 < NB ? kbk + 2 : kbk) * 32 * HDIM_ + tid * 8, 0, 1);
      WAIT_ASYNC(2);                       // current block's 2 copies done (in-order)
    } else {
      WAIT_ASYNC(0);
    }
    __syncthreads();                       // all waves' copies for this block landed

    const _Float16* kBuf = kS[kbk & 1];
    const _Float16* vBuf = vS[kbk & 1];

    // scores S[16 x 32] = Q(16x64) . K^T : two 16-col tiles, two d-chunks each
    v8f s0 = {}, s1 = {};
    s0 = wmma16(qa0, load_b_frag(kBuf, 64, 0, 0),  s0);
    s0 = wmma16(qa1, load_b_frag(kBuf, 64, 32, 0), s0);
    s1 = wmma16(qa0, load_b_frag(kBuf, 64, 0, 16), s1);
    s1 = wmma16(qa1, load_b_frag(kBuf, 64, 32, 16), s1);

    // online softmax (rows split: vgpr r -> row r + 8*hi; reduce across 16-lane halves)
    float alpha[8];
#pragma unroll
    for (int r = 0; r < 8; ++r) {
      float a0 = s0[r] * SCALE_, a1 = s1[r] * SCALE_;
      float mx = fmaxf(a0, a1);
#pragma unroll
      for (int off = 8; off > 0; off >>= 1) mx = fmaxf(mx, __shfl_xor(mx, off, 16));
      const float mnew = fmaxf(m[r], mx);
      alpha[r] = __expf(m[r] - mnew);
      m[r] = mnew;
      const float p0 = __expf(a0 - mnew), p1 = __expf(a1 - mnew);
      s0[r] = p0; s1[r] = p1;
      float rs = p0 + p1;
#pragma unroll
      for (int off = 8; off > 0; off >>= 1) rs += __shfl_xor(rs, off, 16);
      l[r] = l[r] * alpha[r] + rs;
    }
#pragma unroll
    for (int r = 0; r < 8; ++r) { o0t[r] *= alpha[r]; o1t[r] *= alpha[r]; o2t[r] *= alpha[r]; o3t[r] *= alpha[r]; }

    // P (C-layout) -> per-wave LDS -> A-layout fragment
    store_c_f16(pw,      32, s0);
    store_c_f16(pw + 16, 32, s1);
    const v16h pa = load_a_frag(pw, 32, 0, 0);

    // O += P(16x32) . V(32x64) : four 16-col d tiles
    o0t = wmma16(pa, load_b_frag(vBuf, 32, 0, 0),  o0t);
    o1t = wmma16(pa, load_b_frag(vBuf, 32, 0, 16), o1t);
    o2t = wmma16(pa, load_b_frag(vBuf, 32, 0, 32), o2t);
    o3t = wmma16(pa, load_b_frag(vBuf, 32, 0, 48), o3t);

    __syncthreads();   // everyone done reading this buffer before it is re-staged
  }

  // epilogue: O / l -> obuf[b, n, h*64 + d] f16 (K-contiguous for out-proj B operand)
  const int col = lane & 15, hi = lane >> 4;
#pragma unroll
  for (int r = 0; r < 8; ++r) {
    const int row = q0 + r + 8 * hi;
    const float inv_l = 1.0f / l[r];
    _Float16* dst = obuf + ((size_t)(b * N_ + row)) * INNER_ + h * HDIM_;
    dst[0  + col] = (_Float16)(o0t[r] * inv_l);
    dst[16 + col] = (_Float16)(o1t[r] * inv_l);
    dst[32 + col] = (_Float16)(o2t[r] * inv_l);
    dst[48 + col] = (_Float16)(o3t[r] * inv_l);
  }
}

// ---------------- Kernel 6: output projection + residual ----------------
// y[b,co,n] = sum_c2 w_out[co,c2] * obuf[b,n,c2] + b_out[co] + x[b,co,n]
__global__ void out_proj_kernel(const _Float16* __restrict__ wout_h, const _Float16* __restrict__ obuf,
                                const float* __restrict__ b_out, const float* __restrict__ x,
                                float* __restrict__ y) {
  const int wave = threadIdx.x >> 5;
  const int lane = threadIdx.x & 31;
  const int b   = blockIdx.z;
  const int co0 = blockIdx.x * 16;
  const int n0  = blockIdx.y * 128 + wave * 16;
  const _Float16* ob = obuf + (size_t)b * N_ * INNER_;

  v8f c = {};
#pragma unroll
  for (int kk = 0; kk < 16; ++kk)
    c = wmma16(load_a_frag(wout_h, INNER_, co0, kk * 32), load_b_frag(ob, INNER_, kk * 32, n0), c);

  const int col = lane & 15, hi = lane >> 4;
#pragma unroll
  for (int r = 0; r < 8; ++r) {
    const int row = co0 + r + 8 * hi;
    const int n = n0 + col;
    const size_t idx = ((size_t)(b * C_ + row)) * N_ + n;
    y[idx] = c[r] + b_out[row] + x[idx];
  }
}

extern "C" void kernel_launch(void* const* d_in, const int* in_sizes, int n_in,
                              void* d_out, int out_size, void* d_ws, size_t ws_size,
                              hipStream_t stream) {
  (void)in_sizes; (void)n_in; (void)out_size; (void)ws_size;
  const float* x      = (const float*)d_in[0];
  const float* gamma  = (const float*)d_in[1];
  const float* beta   = (const float*)d_in[2];
  const float* w_qkv  = (const float*)d_in[3];
  const float* b_qkv  = (const float*)d_in[4];
  const float* w_out  = (const float*)d_in[5];
  const float* b_out  = (const float*)d_in[6];
  float* y = (float*)d_out;

  char* ws = (char*)d_ws;
  size_t off = 0;
  auto take = [&](size_t bytes) { void* p = ws + off; off += (bytes + 255) & ~(size_t)255; return p; };
  float*     stats  = (float*)    take(32 * sizeof(float));
  _Float16*  w_h    = (_Float16*) take((size_t)QKV_ROWS_ * C_ * 2);      // 768 KB
  _Float16*  wout_h = (_Float16*) take((size_t)C_ * INNER_ * 2);        // 256 KB
  _Float16*  xnT    = (_Float16*) take((size_t)B_ * N_ * C_ * 2);       // 4 MB
  _Float16*  qbuf   = (_Float16*) take((size_t)B_ * HEADS_ * N_ * HDIM_ * 2); // 8 MB
  _Float16*  kbuf   = (_Float16*) take((size_t)B_ * HEADS_ * N_ * HDIM_ * 2); // 8 MB
  _Float16*  vT     = (_Float16*) take((size_t)B_ * HEADS_ * HDIM_ * N_ * 2); // 8 MB
  _Float16*  obuf   = (_Float16*) take((size_t)B_ * N_ * INNER_ * 2);   // 8 MB

  gn_stats_kernel<<<16, 256, 0, stream>>>(x, stats);
  gn_apply_kernel<<<B_ * N_, 256, 0, stream>>>(x, gamma, beta, stats, xnT);
  convert_w_kernel<<<(QKV_ROWS_ * C_ + C_ * INNER_ + 255) / 256, 256, 0, stream>>>(w_qkv, w_out, w_h, wout_h);
  qkv_gemm_kernel<<<dim3(QKV_ROWS_ / 16, N_ / 128, B_), 256, 0, stream>>>(w_h, xnT, b_qkv, qbuf, kbuf, vT);
  attention_kernel<<<dim3(N_ / 128, B_ * HEADS_), 256, 0, stream>>>(qbuf, kbuf, vT, obuf);
  out_proj_kernel<<<dim3(C_ / 16, N_ / 128, B_), 256, 0, stream>>>(wout_h, obuf, b_out, x, y);
}
